// Dynamic_GAT_24232205484264
// MI455X (gfx1250) — compile-verified, matrix operand order checked by reading, not run on hardware
//
#include <hip/hip_runtime.h>

// ---------------- types ----------------
typedef __attribute__((ext_vector_type(16))) _Float16 v16h;
typedef __attribute__((ext_vector_type(8)))  _Float16 v8h;
typedef __attribute__((ext_vector_type(8)))  float    v8f;

#define B_  32
#define N_  513
#define D_  768
#define H_  6
#define DH_ 128
#define BN_ (B_ * N_)          // 16416
#define MT_ (BN_ / 16)         // 1026 m-tiles (exact)

// K-index mapping for the 16-bit A/B fragment layout (wave32):
// lane<16: elems 0..7 -> K 0..7, elems 8..15 -> K 16..23
// lane>=16: elems 0..7 -> K 8..15, elems 8..15 -> K 24..31
__device__ __forceinline__ int kmap(int e, int kh) {
    return ((e < 8) ? e : (e + 8)) + 8 * kh;
}

// ---------------- elementwise convert ----------------
__global__ void k_f32_to_f16(const float* __restrict__ s, _Float16* __restrict__ d, long n) {
    long i = (long)blockIdx.x * blockDim.x + threadIdx.x;
    long stride = (long)gridDim.x * blockDim.x;
    for (; i < n; i += stride) d[i] = (_Float16)s[i];
}

__global__ void k_write_l0(float* p) { p[0] = 0.0f; }

// ---------------- sparse stats (one block per batch) ----------------
__global__ void k_stats(const int* __restrict__ adj, const float* __restrict__ pmask,
                        float* __restrict__ c_sp, float* __restrict__ g_sp, int layer) {
    int b = blockIdx.x;
    const int* A = adj + (size_t)b * N_ * N_;
    float ps = 0.f, cs = 0.f, gs = 0.f;
    for (int i = threadIdx.x; i < N_; i += 256) ps += pmask[b * N_ + i];
    for (int k = threadIdx.x; k < N_ - 1; k += 256) cs += (float)A[(N_ - 1) * N_ + k];
    for (int idx = threadIdx.x; idx < 512 * 512; idx += 256) {
        int q = idx >> 9, k = idx & 511;
        gs += (float)A[q * N_ + k];
    }
    __shared__ float sp[256], sc[256], sg[256];
    sp[threadIdx.x] = ps; sc[threadIdx.x] = cs; sg[threadIdx.x] = gs;
    __syncthreads();
    for (int s = 128; s > 0; s >>= 1) {
        if (threadIdx.x < s) {
            sp[threadIdx.x] += sp[threadIdx.x + s];
            sc[threadIdx.x] += sc[threadIdx.x + s];
            sg[threadIdx.x] += sg[threadIdx.x + s];
        }
        __syncthreads();
    }
    if (threadIdx.x == 0) {
        float P = sp[0];
        c_sp[b * 2 + layer] = sc[0] / P;
        g_sp[b * 2 + layer] = sg[0] / (P * P);
    }
}

// ---------------- qk = featH (16416x768) x WhT (768x768), f16 in / f32 acc ----------------
// 8 waves per WG; wave -> one 16x16 D tile. B[k,n] = W[n,k] (W row-major) so B-fragment
// columns are contiguous rows of W -> two b128 loads per lane per K-step, same as A.
__global__ void k_gemm_qk(const _Float16* __restrict__ F, const _Float16* __restrict__ W,
                          float* __restrict__ qkF, _Float16* __restrict__ qkH) {
    const int lane = threadIdx.x & 31;
    const int wave = threadIdx.x >> 5;
    const int mtile = blockIdx.y * 8 + wave;
    if (mtile >= MT_) return;           // uniform per wave, no barriers in kernel
    const int ntile = blockIdx.x;
    const int r  = lane & 15;
    const int kh = lane >> 4;

    const _Float16* Arow = F + (size_t)(mtile * 16 + r) * D_;
    const _Float16* Brow = W + (size_t)(ntile * 16 + r) * D_;   // B col n = W row n

    v8f acc = {};
    for (int k0 = 0; k0 < D_; k0 += 32) {
        v8h a0 = *(const v8h*)(Arow + k0 + 8 * kh);
        v8h a1 = *(const v8h*)(Arow + k0 + 16 + 8 * kh);
        v8h b0 = *(const v8h*)(Brow + k0 + 8 * kh);
        v8h b1 = *(const v8h*)(Brow + k0 + 16 + 8 * kh);
        v16h a, bm;
#pragma unroll
        for (int i = 0; i < 8; ++i) { a[i] = a0[i]; a[i + 8] = a1[i]; bm[i] = b0[i]; bm[i + 8] = b1[i]; }
        acc = __builtin_amdgcn_wmma_f32_16x16x32_f16(false, a, false, bm, (short)0, acc, false, false);
    }
    // D layout: VGPR j -> row (j + 8*kh), col = lane&15
    const int col = ntile * 16 + r;
    size_t base = (size_t)(mtile * 16 + 8 * kh) * D_ + col;
#pragma unroll
    for (int j = 0; j < 8; ++j) {
        float v = acc[j];
        qkF[base + (size_t)j * D_] = v;
        qkH[base + (size_t)j * D_] = (_Float16)v;
    }
}

// ---------------- sq/sk: one wave per (b,n,h) ----------------
__global__ void k_sqsk(const float* __restrict__ qkF, const float* __restrict__ attw,
                       float* __restrict__ sq, float* __restrict__ sk) {
    int gw = blockIdx.x * 8 + (threadIdx.x >> 5);
    int lane = threadIdx.x & 31;
    if (gw >= BN_ * H_) return;
    int h = gw % H_;
    int bn = gw / H_;
    const float* row = qkF + (size_t)bn * D_ + h * DH_;
    const float* wq = attw + h * (2 * DH_);
    const float* wk = wq + DH_;
    float aq = 0.f, ak = 0.f;
    for (int i = lane; i < DH_; i += 32) {
        float v = row[i];
        aq += v * wq[i];
        ak += v * wk[i];
    }
    for (int off = 16; off; off >>= 1) {
        aq += __shfl_xor(aq, off);
        ak += __shfl_xor(ak, off);
    }
    if (lane == 0) { sq[gw] = aq; sk[gw] = ak; }
}

// ---------------- fused attention: grid (33 qtiles, 6 heads, 32 batches) ----------------
// tanh is bounded -> exp in [e^-1, e] -> single-pass softmax (no running max).
// 8 waves each own 16 d-columns; V tile (32x128 f16) staged in LDS per K-step.
__global__ void k_attn(const float* __restrict__ sq, const float* __restrict__ sk,
                       const int* __restrict__ adj, const _Float16* __restrict__ qkH,
                       const float* __restrict__ featIn, float* __restrict__ featOut) {
    __shared__ __align__(16) _Float16 Vt[32 * DH_];
    __shared__ float rsum[16];

    const int qt = blockIdx.x;
    const int h  = blockIdx.y;
    const int b  = blockIdx.z;
    const int lane = threadIdx.x & 31;
    const int wave = threadIdx.x >> 5;
    const int r  = lane & 15;
    const int kh = lane >> 4;
    const int dbase = wave * 16;

    const int q  = qt * 16 + r;
    const int qc = (q < N_) ? q : (N_ - 1);
    const float sqv = sq[(b * N_ + qc) * H_ + h];
    const int* Ab = adj + ((size_t)b * N_ + qc) * N_;

    v8f acc = {};
    float rs = 0.f;

    for (int k0 = 0; k0 < N_; k0 += 32) {
        __syncthreads();
        {   // cooperative V-tile load (coalesced b128s)
            int kk = threadIdx.x >> 3;
            int dp = (threadIdx.x & 7) * 16;
            int kg = k0 + kk;
            v8h z = {}, x0 = z, x1 = z;
            if (kg < N_) {
                const _Float16* src = qkH + (size_t)(b * N_ + kg) * D_ + h * DH_ + dp;
                x0 = *(const v8h*)src;
                x1 = *(const v8h*)(src + 8);
            }
            *(v8h*)(&Vt[kk * DH_ + dp])     = x0;
            *(v8h*)(&Vt[kk * DH_ + dp + 8]) = x1;
        }
        __syncthreads();

        v16h P, Bv;
#pragma unroll
        for (int e = 0; e < 16; ++e) {
            int krel = kmap(e, kh);
            int k = k0 + krel;
            float p = 0.f;
            if (k < N_) {
                float s = tanhf(sqv + sk[(b * N_ + k) * H_ + h]);
                if (Ab[k] <= 0) s = -1e-8f;
                p = __expf(s);
            }
            P[e] = (_Float16)p;
            rs += p;
            Bv[e] = Vt[krel * DH_ + dbase + r];
        }
        acc = __builtin_amdgcn_wmma_f32_16x16x32_f16(false, P, false, Bv, (short)0, acc, false, false);
    }

    rs += __shfl_xor(rs, 16);                 // combine the two K-halves of each row
    if (threadIdx.x < 16) rsum[threadIdx.x] = rs;   // all waves agree; wave0 publishes
    __syncthreads();

#pragma unroll
    for (int j = 0; j < 8; ++j) {
        int rt = j + 8 * kh;
        int qg = qt * 16 + rt;
        if (qg < N_) {
            float denom = rsum[rt];
            size_t idx = (size_t)(b * N_ + qg) * D_ + h * DH_ + dbase + r;
            featOut[idx] = featIn[idx] + tanhf(acc[j] / denom);
        }
    }
}

// ---------------- host ----------------
extern "C" void kernel_launch(void* const* d_in, const int* in_sizes, int n_in,
                              void* d_out, int out_size, void* d_ws, size_t ws_size,
                              hipStream_t stream) {
    (void)in_sizes; (void)n_in; (void)out_size; (void)ws_size;
    const float* pmask   = (const float*)d_in[0];
    const float* feature = (const float*)d_in[1];
    const int*   adjs    = (const int*)d_in[2];
    const float* W       = (const float*)d_in[3];
    const float* attw    = (const float*)d_in[4];

    float* outF = (float*)d_out;                 // (B,N,D) feature
    float* l0   = outF + (size_t)BN_ * D_;       // scalar 0.0
    float* c_sp = l0 + 1;                        // (B,2)
    float* g_sp = c_sp + B_ * 2;                 // (B,2)

    char* ws = (char*)d_ws;
    size_t o = 0;
    auto carve = [&](size_t bytes) -> void* {
        void* p = ws + o;
        o = (o + bytes + 255) & ~(size_t)255;
        return p;
    };
    _Float16* Wh    = (_Float16*)carve((size_t)D_ * D_ * 2);
    _Float16* featH = (_Float16*)carve((size_t)BN_ * D_ * 2);
    float*    qkF   = (float*)   carve((size_t)BN_ * D_ * 4);
    _Float16* qkH   = (_Float16*)carve((size_t)BN_ * D_ * 2);
    float*    sqB   = (float*)   carve((size_t)BN_ * H_ * 4);
    float*    skB   = (float*)   carve((size_t)BN_ * H_ * 4);
    float*    feat1 = (float*)   carve((size_t)BN_ * D_ * 4);

    k_f32_to_f16<<<512, 256, 0, stream>>>(W, Wh, (long)D_ * D_);
    k_write_l0<<<1, 1, 0, stream>>>(l0);

    const float* fin = feature;
    for (int l = 0; l < 2; ++l) {
        const int* adjL = adjs + (size_t)l * B_ * N_ * N_;
        k_stats<<<B_, 256, 0, stream>>>(adjL, pmask, c_sp, g_sp, l);
        k_f32_to_f16<<<2048, 256, 0, stream>>>(fin, featH, (long)BN_ * D_);
        k_gemm_qk<<<dim3(D_ / 16, (MT_ + 7) / 8), 256, 0, stream>>>(featH, Wh, qkF, qkH);
        k_sqsk<<<(BN_ * H_) / 8, 256, 0, stream>>>(qkF, attw, sqB, skB);
        float* fout = (l == 0) ? feat1 : outF;
        k_attn<<<dim3((N_ + 15) / 16, H_, B_), 256, 0, stream>>>(sqB, skB, adjL, qkH, fin, fout);
        fin = feat1;
    }
}
